// SimCLRNet_14439680049650
// MI455X (gfx1250) — matrix operand order repaired and checked by reading, not compiled, not run on hardware
//
#include <hip/hip_runtime.h>
#include <hip/hip_bf16.h>

typedef _Float16 v16h __attribute__((ext_vector_type(16)));
typedef float    v8f  __attribute__((ext_vector_type(8)));
typedef _Float16 h4   __attribute__((ext_vector_type(4)));

#define NB 8192     // batch
#define NN 16384    // 2*NB
#define DD 128      // feature dim
#define LDSPITCH 136   // 128 + 8 f16 pad -> 272B row stride, conflict-free b128 reads
#define SCALE (1.0f / 0.07f)

union FragU { uint4 u[2]; v16h v; };

// A-fragment (16x32 f16, MxK): lane l (lr=l%16, half=l/16) holds row M=lr:
//   v[0..7]  = K = k0 + 8*half + 0..7
//   v[8..15] = K = k0 + 16 + 8*half + 0..7
__device__ __forceinline__ v16h loadA(const _Float16 s[][LDSPITCH], int row, int half, int k0) {
    FragU f;
    const uint4* p = (const uint4*)&s[row][k0 + 8 * half];
    f.u[0] = p[0];      // 8 f16 at k0+8*half
    f.u[1] = p[2];      // 8 f16 at k0+16+8*half  (+32 bytes)
    return f.v;
}

// B-fragment (32x16 f16, KxN): lane l holds column N=lr:
//   v[0..7]  = K = k0 + 8*half + 0..7, v[8..15] = K = k0 + 16 + 8*half + 0..7
__device__ __forceinline__ v16h loadB(const _Float16 s[][LDSPITCH], int row, int half, int k0) {
    FragU f;
    const uint4* p = (const uint4*)&s[row][k0 + 8 * half];
    f.u[0] = p[0];
    f.u[1] = p[2];
    return f.v;
}

// One wave per row: normalize and split into f16 hi/lo planes.
__global__ void nrm_split(const float* __restrict__ q, const float* __restrict__ k,
                          _Float16* __restrict__ hi, _Float16* __restrict__ lo) {
    int w = threadIdx.x >> 5, lane = threadIdx.x & 31;
    int row = blockIdx.x * 8 + w;
    const float4* src = (row < NB) ? ((const float4*)q + (size_t)row * (DD / 4))
                                   : ((const float4*)k + (size_t)(row - NB) * (DD / 4));
    float4 v = src[lane];
    float s = v.x * v.x + v.y * v.y + v.z * v.z + v.w * v.w;
#pragma unroll
    for (int off = 16; off > 0; off >>= 1) s += __shfl_xor(s, off, 32);
    float inv = 1.0f / fmaxf(sqrtf(s), 1e-12f);

    float xs[4] = {v.x, v.y, v.z, v.w};
    h4 hv, lv;
#pragma unroll
    for (int e = 0; e < 4; ++e) {
        float xn = xs[e] * inv;
        _Float16 h = (_Float16)xn;          // round-to-nearest high part
        _Float16 l = (_Float16)(xn - (float)h);  // residual low part
        hv[e] = h; lv[e] = l;
    }
    ((h4*)hi)[(size_t)row * (DD / 4) + lane] = hv;
    ((h4*)lo)[(size_t)row * (DD / 4) + lane] = lv;
}

__global__ void labels_zero(unsigned* __restrict__ out) {
    out[blockIdx.x * 256 + threadIdx.x] = 0u;
}

// 128x128 output tile per block; 8 waves, each wave owns a 16x128 strip.
__global__ void __launch_bounds__(256)
simclr_gemm(const _Float16* __restrict__ hi, const _Float16* __restrict__ lo,
            float* __restrict__ out) {
    __shared__ __align__(16) _Float16 sAhi[128][LDSPITCH];
    __shared__ __align__(16) _Float16 sAlo[128][LDSPITCH];
    __shared__ __align__(16) _Float16 sBhi[128][LDSPITCH];
    __shared__ __align__(16) _Float16 sBlo[128][LDSPITCH];

    const int t = threadIdx.x;
    const int rowBase = blockIdx.y * 128;
    const int colBase = blockIdx.x * 128;

    // Stage full-K rows for both sides (f data is L2-resident; 16B chunks coalesce).
    const uint4* gHi = (const uint4*)hi;
    const uint4* gLo = (const uint4*)lo;
#pragma unroll
    for (int it = 0; it < 8; ++it) {
        int c = it * 256 + t;
        int r = c >> 4;          // row 0..127
        int kc = c & 15;         // 16B chunk 0..15 (row = 256B = 16 uint4)
        *(uint4*)&sAhi[r][kc * 8] = gHi[(size_t)(rowBase + r) * 16 + kc];
        *(uint4*)&sAlo[r][kc * 8] = gLo[(size_t)(rowBase + r) * 16 + kc];
        *(uint4*)&sBhi[r][kc * 8] = gHi[(size_t)(colBase + r) * 16 + kc];
        *(uint4*)&sBlo[r][kc * 8] = gLo[(size_t)(colBase + r) * 16 + kc];
    }
    __syncthreads();

    const int w = t >> 5, lane = t & 31;
    const int half = lane >> 4, lr = lane & 15;

    v8f acc[8];
#pragma unroll
    for (int ni = 0; ni < 8; ++ni) acc[ni] = (v8f){0, 0, 0, 0, 0, 0, 0, 0};

#pragma unroll
    for (int k0 = 0; k0 < DD; k0 += 32) {
        v16h aH = loadA(sAhi, w * 16 + lr, half, k0);
        v16h aL = loadA(sAlo, w * 16 + lr, half, k0);
#pragma unroll
        for (int ni = 0; ni < 8; ++ni) {
            v16h bH = loadB(sBhi, ni * 16 + lr, half, k0);
            v16h bL = loadB(sBlo, ni * 16 + lr, half, k0);
            // x*y ~= hi*hi + hi*lo + lo*hi, f32-accumulated
            acc[ni] = __builtin_amdgcn_wmma_f32_16x16x32_f16(
                false, aH, false, bH, (short)0, acc[ni], false, false);
            acc[ni] = __builtin_amdgcn_wmma_f32_16x16x32_f16(
                false, aH, false, bL, (short)0, acc[ni], false, false);
            acc[ni] = __builtin_amdgcn_wmma_f32_16x16x32_f16(
                false, aL, false, bH, (short)0, acc[ni], false, false);
        }
    }

    // Scatter epilogue: logits[i] = [ sim[i, i^NB] | sim[i, j!=i,p] ] / TEMP
    // C-frag layout: VGPR r -> row M = r + 8*half, col N = lr.
#pragma unroll
    for (int ni = 0; ni < 8; ++ni) {
        int j = colBase + ni * 16 + lr;
#pragma unroll
        for (int r = 0; r < 8; ++r) {
            int i = rowBase + w * 16 + r + 8 * half;
            if (j == i) continue;                       // drop diagonal
            int a = i & (NB - 1);                       // min(i, p)
            int b2 = a + NB;                            // max(i, p)
            int dest = (j == (i ^ NB))
                           ? 0                          // positive pair -> col 0
                           : (1 + j - (j > a) - (j > b2));
            __builtin_nontemporal_store(acc[ni][r] * SCALE,
                                        &out[(size_t)i * (NN - 1) + dest]);
        }
    }
}

extern "C" void kernel_launch(void* const* d_in, const int* in_sizes, int n_in,
                              void* d_out, int out_size, void* d_ws, size_t ws_size,
                              hipStream_t stream) {
    const float* q = (const float*)d_in[0];
    const float* k = (const float*)d_in[1];
    float* out = (float*)d_out;

    _Float16* hi = (_Float16*)d_ws;                 // 4 MB
    _Float16* lo = hi + (size_t)NN * DD;            // 4 MB

    nrm_split<<<NN / 8, 256, 0, stream>>>(q, k, hi, lo);
    labels_zero<<<NN / 256, 256, 0, stream>>>(
        (unsigned*)(out + (size_t)NN * (NN - 1)));
    simclr_gemm<<<dim3(NN / 128, NN / 128), 256, 0, stream>>>(hi, lo, out);
}
